// DynamicMixture_17042430230997
// MI455X (gfx1250) — compile-verified, matrix operand order checked by reading, not run on hardware
//
#include <hip/hip_runtime.h>
#include <hip/hip_bf16.h>

// Problem constants (from reference)
#define T_STEPS 2048
#define B_DIM   256
#define F_DIM   64
#define K_MIX   8
#define ESN_DIM 512
#define HID_DIM 256
#define LOG2PI  1.8378770664093453f

typedef __attribute__((ext_vector_type(16))) __bf16 v16bf;
typedef __attribute__((ext_vector_type(8)))  float  v8f;

// ---------------------------------------------------------------------------
// WMMA helpers (CDNA5 gfx1250, wave32)
// ---------------------------------------------------------------------------
__device__ __forceinline__ v8f wmma_bf16(v16bf a, v16bf b, v8f c) {
  // D = A(16x32 bf16) x B(32x16 bf16) + C(16x16 f32)
  return __builtin_amdgcn_wmma_f32_16x16x32_bf16(
      /*neg_a=*/false, a, /*neg_b=*/false, b,
      /*c_mod=*/(short)0, c, /*reuse_a=*/false, /*reuse_b=*/false);
}

// Load one 16x32 bf16 operand fragment.
// A operand: base -> element [row 0][k0] of a row-major tile, stride in halves.
// Per documented layout: lane L holds row (L&15); VGPR halves cover
// K = (i<8?0:16) + ((L>=16)?8:0) + (i&7)  -> two contiguous 16B chunks.
// B operand uses the mirrored layout (lane -> N column), so the same loader
// works with base = &W[n0][k0] and stride = K_total (weights are [N,K] row-major).
__device__ __forceinline__ v16bf load_frag(const __bf16* base, int strideHalves, int lane) {
  const __bf16* p = base + (lane & 15) * strideHalves + ((lane >> 4) & 1) * 8;
  v16bf a;
  uint4* u = (uint4*)&a;
  u[0] = *(const uint4*)(p);
  u[1] = *(const uint4*)(p + 16);
  return a;
}

__device__ __forceinline__ float tanh_fast(float x) {
  float e = __expf(2.0f * x);          // saturates cleanly at +/-inf
  return 1.0f - 2.0f / (e + 1.0f);
}

__device__ __forceinline__ float softplus_f(float x) {
  return (x > 20.0f) ? x : __logf(1.0f + __expf(x));
}

// ---------------------------------------------------------------------------
// Weight f32 -> bf16 conversion (once, into workspace; weights then live in L2)
// ---------------------------------------------------------------------------
__global__ void cvt_f32_bf16(const float* __restrict__ src, __bf16* __restrict__ dst, int n) {
  int i = blockIdx.x * blockDim.x + threadIdx.x;
  if (i < n) dst[i] = (__bf16)src[i];
}

// Zero-pad Ww [K=8, HID] to [16, HID] bf16 so the tiny head can use WMMA (N=16 min).
__global__ void cvt_ww_pad(const float* __restrict__ src, __bf16* __restrict__ dst) {
  int i = blockIdx.x * blockDim.x + threadIdx.x;  // 16*HID
  if (i < 16 * HID_DIM) {
    int r = i / HID_DIM;
    dst[i] = (r < K_MIX) ? (__bf16)src[i] : (__bf16)0.0f;
  }
}

// ---------------------------------------------------------------------------
// Persistent recurrence kernel: 16 workgroups, each owns a 16-row batch tile
// and runs all T steps with LDS-resident state.
// ---------------------------------------------------------------------------
__global__ void __launch_bounds__(256, 1)
dyn_mixture_scan(const float* __restrict__ x,       // [T,B,F] f32
                 const int*   __restrict__ seqlen,  // [B]
                 const float* __restrict__ b1,      // [HID]
                 const float* __restrict__ bw,      // [K]
                 const float* __restrict__ bm,      // [K*F]
                 const float* __restrict__ bc,      // [K*F]
                 const __bf16* __restrict__ W1b,    // [HID,ESN]   bf16
                 const __bf16* __restrict__ Wmb,    // [K*F,HID]   bf16
                 const __bf16* __restrict__ Wcb,    // [K*F,HID]   bf16
                 const __bf16* __restrict__ Winb,   // [ESN,F]     bf16
                 const __bf16* __restrict__ Wresb,  // [ESN,ESN]   bf16
                 const __bf16* __restrict__ Wwb,    // [16,HID]    bf16 (zero-padded)
                 float* __restrict__ out)           // [B]
{
  // LDS state (padded strides keep per-lane b128 reads 16B aligned, bank-spread)
  __shared__ __align__(16) __bf16 h_buf[2][16 * 520];  // h state, double buffered
  __shared__ __align__(16) __bf16 hid_s[16 * 264];     // tanh hidden
  __shared__ __align__(16) float  x_f[16 * 64];        // x_t in f32 (for z)
  __shared__ __align__(16) __bf16 x_bf[16 * 72];       // x_t in bf16 (for GEMM)
  __shared__ float comp_lp[16 * 8];                    // per (row,k) gaussian logp
  __shared__ float a_w[16 * 8];                        // mixture-weight logits
  __shared__ float acc_s[16];                          // running log-likelihood
  __shared__ float b1_s[HID_DIM], bm_s[K_MIX * F_DIM], bc_s[K_MIX * F_DIM], bw_s[K_MIX];

  const int tid  = threadIdx.x;
  const int lane = tid & 31;
  const int w    = tid >> 5;           // 8 waves
  const int m0   = blockIdx.x * 16;    // batch tile base

  // ---- init ----
  for (int i = tid; i < 2 * 16 * 520; i += 256) ((__bf16*)h_buf)[i] = (__bf16)0.0f;
  if (tid < HID_DIM) b1_s[tid] = b1[tid];
  for (int i = tid; i < K_MIX * F_DIM; i += 256) { bm_s[i] = bm[i]; bc_s[i] = bc[i]; }
  if (tid < K_MIX) bw_s[tid] = bw[tid];
  if (tid < 128) comp_lp[tid] = 0.0f;
  if (tid < 16)  acc_s[tid]  = 0.0f;
  __syncthreads();

  for (int t = 0; t < T_STEPS; ++t) {
    __bf16* h_cur = h_buf[t & 1];
    __bf16* h_nxt = h_buf[(t + 1) & 1];

    // ---- phase X: stage x_t (f32 + bf16 copies), prefetch next step ----
    {
      int m = tid >> 4;
      int f = (tid & 15) << 2;
      size_t g = ((size_t)t * B_DIM + (m0 + m)) * F_DIM + f;
      float4 v = *(const float4*)(x + g);
      *(float4*)(&x_f[m * 64 + f]) = v;
      x_bf[m * 72 + f + 0] = (__bf16)v.x;
      x_bf[m * 72 + f + 1] = (__bf16)v.y;
      x_bf[m * 72 + f + 2] = (__bf16)v.z;
      x_bf[m * 72 + f + 3] = (__bf16)v.w;
      if (t + 1 < T_STEPS)
        __builtin_prefetch(x + g + (size_t)B_DIM * F_DIM, 0, 0);
    }

    // ---- phase A: hid = tanh(h @ W1^T + b1)  [16 x 256], K = 512 ----
    for (int nt = w; nt < 16; nt += 8) {
      int n0 = nt * 16;
      v8f c = {};
      for (int kk = 0; kk < 16; ++kk) {
        v16bf a = load_frag(h_cur + kk * 32, 520, lane);
        v16bf b = load_frag(W1b + n0 * ESN_DIM + kk * 32, ESN_DIM, lane);
        c = wmma_bf16(a, b, c);
      }
      int col = n0 + (lane & 15);
      float bias = b1_s[col];
      int hl = (lane >> 4) & 1;
      for (int r = 0; r < 8; ++r) {
        int m = r + hl * 8;
        hid_s[m * 264 + col] = (__bf16)tanh_fast(c[r] + bias);
      }
    }
    __syncthreads();

    // ---- phase B: means/stds heads + fused gaussian log-prob reduction ----
    // wave w owns mixture component k == w (4 column tiles of 16 = 64 features)
    for (int jt = w * 4; jt < w * 4 + 4; ++jt) {
      int j0 = jt * 16;
      v8f cm = {}, cs = {};
      for (int kk = 0; kk < 8; ++kk) {
        v16bf a   = load_frag(hid_s + kk * 32, 264, lane);
        v16bf b0  = load_frag(Wmb + j0 * HID_DIM + kk * 32, HID_DIM, lane);
        v16bf b1f = load_frag(Wcb + j0 * HID_DIM + kk * 32, HID_DIM, lane);
        cm = wmma_bf16(a, b0, cm);
        cs = wmma_bf16(a, b1f, cs);
      }
      int col = j0 + (lane & 15);
      float biasm = bm_s[col], biasc = bc_s[col];
      int f  = col & 63;
      int k  = col >> 6;              // == w for this wave's tiles
      int hl = (lane >> 4) & 1;
      float partial[8];
      for (int r = 0; r < 8; ++r) {
        int m = r + hl * 8;
        float mean = cm[r] + biasm;
        float sd   = softplus_f(cs[r] + biasc);
        float z    = (x_f[m * 64 + f] - mean) / sd;
        partial[r] = -0.5f * z * z - __logf(sd);
      }
      // reduce across the 16 lanes of each half (sum over feature columns)
      for (int off = 1; off < 16; off <<= 1)
        for (int r = 0; r < 8; ++r)
          partial[r] += __shfl_xor(partial[r], off, 32);
      if ((lane & 15) == 0) {
        for (int r = 0; r < 8; ++r)
          comp_lp[(r + hl * 8) * 8 + k] += partial[r];  // single-owner: race free
      }
    }
    __syncthreads();

    // ---- phase C (wave 0): WMMA weight head + logsumexp + masked accumulate
    // ---- phase D (waves 1-7): h_next = tanh(x @ Win^T + h @ Wres^T)
    if (w == 0) {
      // a_w[16,8] = hid @ Ww^T via zero-padded [16,HID] weights, K = 256
      v8f c = {};
      for (int kk = 0; kk < 8; ++kk) {
        v16bf a = load_frag(hid_s + kk * 32, 264, lane);
        v16bf b = load_frag(Wwb + kk * 32, HID_DIM, lane);
        c = wmma_bf16(a, b, c);
      }
      {
        int n  = lane & 15;            // mixture component index (valid < 8)
        int hl = (lane >> 4) & 1;
        if (n < K_MIX) {
          float bias = bw_s[n];
          for (int r = 0; r < 8; ++r)
            a_w[(r + hl * 8) * 8 + n] = c[r] + bias;
        }
      }
      if (lane < 16) {
        int m = lane;
        float amax = a_w[m * 8];
        for (int k = 1; k < 8; ++k) amax = fmaxf(amax, a_w[m * 8 + k]);
        float sw = 0.0f;
        for (int k = 0; k < 8; ++k) sw += __expf(a_w[m * 8 + k] - amax);
        float lse_w = amax + __logf(sw);
        float tot[8];
        float tmax = -3.4e38f;
        for (int k = 0; k < 8; ++k) {
          tot[k] = (a_w[m * 8 + k] - lse_w) + comp_lp[m * 8 + k]
                   - 0.5f * (float)F_DIM * LOG2PI;
          tmax = fmaxf(tmax, tot[k]);
        }
        float se = 0.0f;
        for (int k = 0; k < 8; ++k) se += __expf(tot[k] - tmax);
        float lp = tmax + __logf(se);
        if (t < seqlen[m0 + m]) acc_s[m] += lp;
        for (int k = 0; k < 8; ++k) comp_lp[m * 8 + k] = 0.0f;  // reset for next step
      }
    } else {
      for (int jt = w - 1; jt < 32; jt += 7) {
        int j0 = jt * 16;
        v8f c = {};
        // x @ Win^T contribution (K = 64)
        for (int kk = 0; kk < 2; ++kk) {
          v16bf a = load_frag(x_bf + kk * 32, 72, lane);
          v16bf b = load_frag(Winb + j0 * F_DIM + kk * 32, F_DIM, lane);
          c = wmma_bf16(a, b, c);
        }
        // h @ Wres^T contribution (K = 512)
        for (int kk = 0; kk < 16; ++kk) {
          v16bf a = load_frag(h_cur + kk * 32, 520, lane);
          v16bf b = load_frag(Wresb + j0 * ESN_DIM + kk * 32, ESN_DIM, lane);
          c = wmma_bf16(a, b, c);
        }
        int col = j0 + (lane & 15);
        int hl = (lane >> 4) & 1;
        for (int r = 0; r < 8; ++r) {
          int m = r + hl * 8;
          h_nxt[m * 520 + col] = (__bf16)tanh_fast(c[r]);
        }
      }
    }
    __syncthreads();  // h_next complete, comp_lp reset, x consumable
  }

  if (tid < 16) out[m0 + tid] = acc_s[tid];
}

// ---------------------------------------------------------------------------
// Host launcher
// ---------------------------------------------------------------------------
extern "C" void kernel_launch(void* const* d_in, const int* in_sizes, int n_in,
                              void* d_out, int out_size, void* d_ws, size_t ws_size,
                              hipStream_t stream) {
  const float* x      = (const float*)d_in[0];
  const int*   seqlen = (const int*)d_in[1];
  const float* W1     = (const float*)d_in[2];
  const float* b1     = (const float*)d_in[3];
  const float* Ww     = (const float*)d_in[4];
  const float* bw     = (const float*)d_in[5];
  const float* Wm     = (const float*)d_in[6];
  const float* bm     = (const float*)d_in[7];
  const float* Wc     = (const float*)d_in[8];
  const float* bc     = (const float*)d_in[9];
  const float* Win    = (const float*)d_in[10];
  const float* Wres   = (const float*)d_in[11];
  float* out = (float*)d_out;

  // bf16 weight cache in workspace (stays L2-resident: ~1.33 MB total)
  __bf16* W1b   = (__bf16*)d_ws;                   // 256*512
  __bf16* Wmb   = W1b   + HID_DIM * ESN_DIM;       // 512*256
  __bf16* Wcb   = Wmb   + K_MIX * F_DIM * HID_DIM; // 512*256
  __bf16* Winb  = Wcb   + K_MIX * F_DIM * HID_DIM; // 512*64
  __bf16* Wresb = Winb  + ESN_DIM * F_DIM;         // 512*512
  __bf16* Wwb   = Wresb + ESN_DIM * ESN_DIM;       // 16*256 zero-padded

  const int blk = 256;
  int n;
  n = HID_DIM * ESN_DIM;       cvt_f32_bf16<<<(n + blk - 1) / blk, blk, 0, stream>>>(W1,   W1b,   n);
  n = K_MIX * F_DIM * HID_DIM; cvt_f32_bf16<<<(n + blk - 1) / blk, blk, 0, stream>>>(Wm,   Wmb,   n);
  n = K_MIX * F_DIM * HID_DIM; cvt_f32_bf16<<<(n + blk - 1) / blk, blk, 0, stream>>>(Wc,   Wcb,   n);
  n = ESN_DIM * F_DIM;         cvt_f32_bf16<<<(n + blk - 1) / blk, blk, 0, stream>>>(Win,  Winb,  n);
  n = ESN_DIM * ESN_DIM;       cvt_f32_bf16<<<(n + blk - 1) / blk, blk, 0, stream>>>(Wres, Wresb, n);
  n = 16 * HID_DIM;            cvt_ww_pad  <<<(n + blk - 1) / blk, blk, 0, stream>>>(Ww, Wwb);

  dyn_mixture_scan<<<B_DIM / 16, 256, 0, stream>>>(
      x, seqlen, b1, bw, bm, bc, W1b, Wmb, Wcb, Winb, Wresb, Wwb, out);
}
